// StructureExtractor_76613626626550
// MI455X (gfx1250) — matrix operand order repaired and checked by reading, not compiled, not run on hardware
//
#include <hip/hip_runtime.h>

#define N_NODES 50000
#define N_NODES_PAD 50048   // padded to a multiple of 64 for the MLP tiles
#define N_EDGES 600000
#define HID 128
#define N_LAYERS 3

// ---------- CDNA5 WMMA types ----------
typedef __attribute__((ext_vector_type(16))) __bf16 v16bf;
typedef __attribute__((ext_vector_type(8)))  float  v8f;

struct __align__(16) U4 { unsigned int x, y, z, w; };
struct __align__(16) P32 { U4 lo, hi; };

__device__ __forceinline__ v8f wmma_bf16(v16bf a, v16bf b, v8f c) {
    // (neg_a, A, neg_b, B, c_mod, C, reuse_a, reuse_b)
    return __builtin_amdgcn_wmma_f32_16x16x32_bf16(false, a, false, b, (short)0, c,
                                                   false, false);
}

// f32 -> bf16 round-to-nearest-even (raw ushort payload)
__device__ __forceinline__ unsigned short f2bf(float f) {
    unsigned int u = __builtin_bit_cast(unsigned int, f);
    u += 0x7FFFu + ((u >> 16) & 1u);
    return (unsigned short)(u >> 16);
}

// load 16 bf16 (as 2x16B) into a WMMA fragment register block
__device__ __forceinline__ v16bf load_frag(const unsigned short* p0,
                                           const unsigned short* p1) {
    P32 t;
    t.lo = *reinterpret_cast<const U4*>(p0);
    t.hi = *reinterpret_cast<const U4*>(p1);
    return __builtin_bit_cast(v16bf, t);
}

__device__ __forceinline__ void atomic_add_f32g(float* p, float v) {
    asm volatile("global_atomic_add_f32 %0, %1, off" : : "v"(p), "v"(v) : "memory");
}

// async copy 16B global -> LDS (ASYNCcnt-tracked)
__device__ __forceinline__ void async_g2l_b128(unsigned int lds_addr, const void* gaddr) {
    asm volatile("global_load_async_to_lds_b128 %0, %1, off"
                 : : "v"(lds_addr), "v"(gaddr) : "memory");
}
__device__ __forceinline__ void wait_asynccnt0() {
    asm volatile("s_wait_asynccnt 0x0" : : : "memory");
}

// ---------- weight prep: f32 [K][N] -> bf16 [N][K] (per layer) ----------
__global__ void prep_weights_kernel(const float* __restrict__ W1,
                                    const float* __restrict__ W2,
                                    unsigned short* __restrict__ W1T,
                                    unsigned short* __restrict__ W2T) {
    int o = blockIdx.x * blockDim.x + threadIdx.x;   // 0 .. 3*128*256-1
    if (o >= N_LAYERS * HID * 2 * HID) return;
    const int per = HID * 2 * HID;                   // 32768
    int l = o / per, r = o % per;
    // W1: [128][256] -> W1T: [256][128]
    {
        int n = r / HID, k = r % HID;
        W1T[o] = f2bf(W1[l * per + k * (2 * HID) + n]);
    }
    // W2: [256][128] -> W2T: [128][256]
    {
        int n = r / (2 * HID), k = r % (2 * HID);
        W2T[o] = f2bf(W2[l * per + k * HID + n]);
    }
}

// ---------- agg = (1+eps[l]) * h ----------
__global__ void init_agg_kernel(const float* __restrict__ h,
                                const float* __restrict__ eps, int l,
                                float* __restrict__ agg) {
    size_t i = (size_t)(blockIdx.x * blockDim.x + threadIdx.x) * 4;
    float s = 1.0f + eps[l];
    const float4 v = *reinterpret_cast<const float4*>(h + i);
    float4 o;
    o.x = s * v.x; o.y = s * v.y; o.z = s * v.z; o.w = s * v.w;
    *reinterpret_cast<float4*>(agg + i) = o;
}

// ---------- agg[dst] += h[src], one wave32 per edge ----------
__global__ void scatter_kernel(const float* __restrict__ h,
                               const int* __restrict__ ei,
                               float* __restrict__ agg) {
    int wid = (blockIdx.x * blockDim.x + threadIdx.x) >> 5;
    int lane = threadIdx.x & 31;
    if (wid >= N_EDGES) return;
    int src = ei[wid];
    int dst = ei[N_EDGES + wid];
    const float4 v =
        *reinterpret_cast<const float4*>(h + (size_t)src * HID + lane * 4);
    float* p = agg + (size_t)dst * HID + lane * 4;
    atomic_add_f32g(p + 0, v.x);
    atomic_add_f32g(p + 1, v.y);
    atomic_add_f32g(p + 2, v.z);
    atomic_add_f32g(p + 3, v.w);
}

// ---------- MLP: relu(relu(z@W1 + b1) @ W2 + b2), 64 nodes per block ----------
// 8 waves; each wave register-blocks 4 M-tiles. W1 staged to LDS via async copy.
__global__ void __launch_bounds__(256) mlp_kernel(
    const float* __restrict__ z,
    const unsigned short* __restrict__ W1T,   // [256][128] bf16
    const float* __restrict__ b1,             // [256]
    const unsigned short* __restrict__ W2T,   // [128][256] bf16
    const float* __restrict__ b2,             // [128]
    float* __restrict__ out) {
    __shared__ __align__(16) unsigned short w1s[2 * HID * HID];    // 256x128 = 64KB
    __shared__ __align__(16) unsigned short zs[64 * HID];          // 64x128  = 16KB
    __shared__ __align__(16) unsigned short interm[64 * 2 * HID];  // 64x256  = 32KB

    const int tid = threadIdx.x;
    const int node0 = blockIdx.x * 64;

    // ---- async-stage W1 (bf16 [256][128]) into LDS: 64KB = 4096 x b128
    {
        unsigned int lds_base = (unsigned int)(uintptr_t)(void*)w1s;
#pragma unroll
        for (int it = 0; it < 16; ++it) {
            int idx = (it * 256 + tid) * 8;          // ushort index, 16B/thread
            async_g2l_b128(lds_base + (unsigned)idx * 2u, W1T + idx);
        }
    }

    // ---- stage 64x128 f32 -> bf16 LDS (32 elems per thread)
#pragma unroll
    for (int it = 0; it < 4; ++it) {
        int i0 = (it * 256 + tid) * 8;
        const float4* p =
            reinterpret_cast<const float4*>(z + (size_t)node0 * HID + i0);
        float4 a = p[0], b = p[1];
        zs[i0 + 0] = f2bf(a.x); zs[i0 + 1] = f2bf(a.y);
        zs[i0 + 2] = f2bf(a.z); zs[i0 + 3] = f2bf(a.w);
        zs[i0 + 4] = f2bf(b.x); zs[i0 + 5] = f2bf(b.y);
        zs[i0 + 6] = f2bf(b.z); zs[i0 + 7] = f2bf(b.w);
    }
    wait_asynccnt0();
    __syncthreads();

    const int wave = tid >> 5;
    const int lane = tid & 31;
    const int row = lane & 15;   // A: M row within tile / B: N column within tile
    const int half = lane >> 4;

    // ---- GEMM1: [64x128] @ [128x256]; wave covers cols [wave*32, wave*32+32)
    v8f acc[4][2] = {};
#pragma unroll
    for (int ks = 0; ks < 4; ++ks) {
        // B 32x16 bf16 from LDS: lanes 0-15 K 0-7 & 16-23; lanes 16-31 K 8-15 & 24-31
        const unsigned short* w0 =
            w1s + (size_t)(wave * 32 + row) * HID + ks * 32 + half * 8;
        const unsigned short* w1 = w0 + 16 * HID;
        v16bf bf0 = load_frag(w0, w0 + 16);
        v16bf bf1 = load_frag(w1, w1 + 16);
#pragma unroll
        for (int m = 0; m < 4; ++m) {
            // A 16x32 bf16: lanes 0-15 K 0-7 & 16-23; lanes 16-31 K 8-15 & 24-31
            v16bf a = load_frag(&zs[(m * 16 + row) * HID + ks * 32 + half * 8],
                                &zs[(m * 16 + row) * HID + ks * 32 + 16 + half * 8]);
            acc[m][0] = wmma_bf16(a, bf0, acc[m][0]);
            acc[m][1] = wmma_bf16(a, bf1, acc[m][1]);
        }
    }
    // bias + ReLU -> bf16 intermediate (C/D: row = r + 8*half, col = lane%16)
    {
        int c0 = wave * 32 + row;
        float bb0 = b1[c0], bb1 = b1[c0 + 16];
#pragma unroll
        for (int m = 0; m < 4; ++m) {
#pragma unroll
            for (int r = 0; r < 8; ++r) {
                int mm = m * 16 + r + half * 8;
                interm[mm * (2 * HID) + c0] = f2bf(fmaxf(acc[m][0][r] + bb0, 0.0f));
                interm[mm * (2 * HID) + c0 + 16] =
                    f2bf(fmaxf(acc[m][1][r] + bb1, 0.0f));
            }
        }
    }
    __syncthreads();

    // ---- GEMM2: [64x256] @ [256x128]; wave covers cols [wave*16, wave*16+16)
    v8f acc2[4] = {};
#pragma unroll
    for (int ks = 0; ks < 8; ++ks) {
        const unsigned short* w =
            W2T + (size_t)(wave * 16 + row) * (2 * HID) + ks * 32 + half * 8;
        v16bf bf = load_frag(w, w + 16);
#pragma unroll
        for (int m = 0; m < 4; ++m) {
            v16bf a =
                load_frag(&interm[(m * 16 + row) * (2 * HID) + ks * 32 + half * 8],
                          &interm[(m * 16 + row) * (2 * HID) + ks * 32 + 16 + half * 8]);
            acc2[m] = wmma_bf16(a, bf, acc2[m]);
        }
    }
    {
        int c = wave * 16 + row;
        float bb = b2[c];
#pragma unroll
        for (int m = 0; m < 4; ++m) {
#pragma unroll
            for (int r = 0; r < 8; ++r) {
                int node = node0 + m * 16 + r + half * 8;
                if (node < N_NODES)
                    out[(size_t)node * HID + c] = fmaxf(acc2[m][r] + bb, 0.0f);
            }
        }
    }
}

extern "C" void kernel_launch(void* const* d_in, const int* in_sizes, int n_in,
                              void* d_out, int out_size, void* d_ws, size_t ws_size,
                              hipStream_t stream) {
    (void)in_sizes; (void)n_in; (void)out_size; (void)ws_size;
    const float* x    = (const float*)d_in[0];
    const int*   edge = (const int*)d_in[1];
    // d_in[2] = batch (unused by the reference computation path)
    const float* eps  = (const float*)d_in[3];
    const float* W1   = (const float*)d_in[4];
    const float* b1   = (const float*)d_in[5];
    const float* W2   = (const float*)d_in[6];
    const float* b2   = (const float*)d_in[7];
    float* out = (float*)d_out;

    const size_t nfeat  = (size_t)N_NODES * HID;       // valid region
    const size_t nfeatp = (size_t)N_NODES_PAD * HID;   // padded buffers
    float* agg = (float*)d_ws;
    float* hA  = agg + nfeatp;
    float* hB  = hA + nfeatp;
    unsigned short* W1T = (unsigned short*)(hB + nfeatp);  // 3*256*128 bf16
    unsigned short* W2T = W1T + N_LAYERS * HID * 2 * HID;

    // weight transpose + bf16 convert
    {
        int total = N_LAYERS * HID * 2 * HID;              // 98304
        prep_weights_kernel<<<(total + 255) / 256, 256, 0, stream>>>(W1, W2, W1T, W2T);
    }

    const float* hin = x;
    for (int l = 0; l < N_LAYERS; ++l) {
        float* hout = (l == N_LAYERS - 1) ? out : ((l & 1) ? hB : hA);

        // agg = (1+eps[l]) * h   (covers the 50000 valid rows)
        init_agg_kernel<<<(int)(nfeat / 4 / 256), 256, 0, stream>>>(hin, eps, l, agg);
        // agg[dst] += h[src]
        scatter_kernel<<<N_EDGES / 8, 256, 0, stream>>>(hin, edge, agg);
        // h = relu(mlp(agg)), 64 nodes per block (tail rows guarded at store)
        mlp_kernel<<<N_NODES_PAD / 64, 256, 0, stream>>>(
            agg, W1T + (size_t)l * HID * 2 * HID, b1 + (size_t)l * 2 * HID,
            W2T + (size_t)l * HID * 2 * HID, b2 + (size_t)l * HID, hout);

        hin = hout;
    }
}
